// SAPAExp_71519795412993
// MI455X (gfx1250) — matrix operand order repaired and checked by reading, not compiled, not run on hardware
//
#include <hip/hip_runtime.h>
#include <hip/hip_bf16.h>

typedef __attribute__((ext_vector_type(16))) _Float16 v16h;
typedef __attribute__((ext_vector_type(8)))  float    v8f;

// ---------------------------------------------------------------------------
// Kernel 1: fused LayerNorm + projection GEMM (used for both q and k paths)
//   dst[p][e] = sum_c LN(src)[p][c] * W[e][c] + bias[e]      (f16 output)
// Block = 128 threads = 4 waves; wave handles 16 pixels x 64 outputs, K=256.
// ---------------------------------------------------------------------------
__global__ __launch_bounds__(128) void ln_proj_kernel(
    const float* __restrict__ src,   // (B, 256, HW) NCHW
    const float* __restrict__ lnw,   // (256)
    const float* __restrict__ lnb,   // (256)
    const float* __restrict__ Wm,    // (64, 256) row-major
    const float* __restrict__ bias,  // (64)
    _Float16* __restrict__ dst,      // (B*HW, 64) f16
    int HW)
{
    __shared__ _Float16 wt[64 * 264];   // weight tile, padded (33.8 KB)
    __shared__ float lw[256];
    __shared__ float lb[256];
    __shared__ float xt[4][16 * 260];   // per-wave pixel tiles, padded (66.5 KB)

    const int t    = threadIdx.x;
    const int wave = t >> 5;
    const int l    = t & 31;
    const int m    = l & 15;            // pixel row within wave tile
    const int hi   = (l >= 16);         // which half-wave

    // --- stage projection weights (f32 -> f16) and LN params, block-wide ---
    #pragma unroll 4
    for (int i = 0; i < 128; ++i) {
        int idx = i * 128 + t;          // 64*256 = 16384 elements
        int e = idx >> 8, c = idx & 255;
        wt[e * 264 + c] = (_Float16)Wm[idx];
    }
    lw[t] = lnw[t]; lw[t + 128] = lnw[t + 128];
    lb[t] = lnb[t]; lb[t + 128] = lnb[t + 128];

    // --- load 16-pixel tile, accumulate per-pixel stats on the fly ---
    const int p0   = blockIdx.x * 64 + wave * 16;    // first pixel of wave tile
    const int b    = p0 / HW;
    const int pin0 = p0 - b * HW;                    // contiguous in NCHW tail dims
    const float* sb = src + (size_t)b * 256 * HW + pin0;

    float sum = 0.f, sq = 0.f;
    float* xrow = &xt[wave][0];
    const int chalf = l >> 4;                        // even/odd channel lanes
    #pragma unroll 4
    for (int i = 0; i < 128; ++i) {
        int c = 2 * i + chalf;
        float v = sb[(size_t)c * HW + m];
        xrow[m * 260 + c] = v;
        sum += v; sq += v * v;
    }
    // lanes l and l^16 hold the two channel-halves of the same pixel
    sum += __shfl_xor(sum, 16, 32);
    sq  += __shfl_xor(sq, 16, 32);
    float mu = sum * (1.0f / 256.0f);
    float rs = rsqrtf(sq * (1.0f / 256.0f) - mu * mu + 1e-5f);
    __syncthreads();

    // --- build A fragments (16x32 f16 per K-step) with LN applied ---
    v16h a[8];
    const int ko = hi ? 16 : 0;
    #pragma unroll
    for (int ks = 0; ks < 8; ++ks) {
        #pragma unroll
        for (int j = 0; j < 8; ++j) {
            int c1 = ks * 32 + ko + j;
            int c2 = c1 + 8;
            a[ks][j]     = (_Float16)((xrow[m * 260 + c1] - mu) * rs * lw[c1] + lb[c1]);
            a[ks][j + 8] = (_Float16)((xrow[m * 260 + c2] - mu) * rs * lw[c2] + lb[c2]);
        }
    }

    // --- 4 N-tiles of 16 outputs, 8 K-steps each ---
    const int n  = l & 15;
    const int bo = hi ? 16 : 0;
    _Float16* drow = dst + (size_t)p0 * 64;
    #pragma unroll
    for (int nt = 0; nt < 4; ++nt) {
        int e = nt * 16 + n;
        float bv = bias[e];
        v8f acc = {bv, bv, bv, bv, bv, bv, bv, bv};
        const _Float16* wr = &wt[e * 264];
        #pragma unroll
        for (int ks = 0; ks < 8; ++ks) {
            v16h bf;
            #pragma unroll
            for (int j = 0; j < 16; ++j) bf[j] = wr[ks * 32 + bo + j];
            acc = __builtin_amdgcn_wmma_f32_16x16x32_f16(
                      false, a[ks], false, bf, (short)0, acc, false, false);
        }
        #pragma unroll
        for (int r = 0; r < 8; ++r) {
            int mm = r + (hi ? 8 : 0);
            drow[(size_t)mm * 64 + e] = (_Float16)acc[r];  // 16 lanes -> 32B rows
        }
    }
}

// ---------------------------------------------------------------------------
// Kernel 2: densified window attention.
// One block = 4x4 low-res cells (16 cells, 64 queries). Union of the 16
// 5x5 windows = 8x8 = 64 pixels. Zero-padded OOB keys/values reproduce the
// reference's zero-pad softmax semantics exactly.
//   sim  = Q(64x64) @ Kunion^T(64x64)      -> WMMA f16, f32 acc
//   attn = masked softmax (25 taps/query)  -> VALU
//   out  = attn(64x64) @ Vunion(64x256)    -> WMMA f16, f32 acc
//   epilogue: transpose through LDS so NCHW stores are 8-float rows
//             (float4 x2 per (c,h_up), 32B-contiguous lane pairs)
// LDS is hand-aliased: the f32 output stage reuses the qls/kls/sim region,
// which is dead after the softmax barrier. Total 113 KB.
// ---------------------------------------------------------------------------
#define OFF_VTL   0                        // 256*72*2 = 36864
#define OFF_ATTN  36864                    // 64*72*2  =  9216
#define OFF_QLS   46080                    // 64*72*2  =  9216  } dead after
#define OFF_KLS   55296                    // 64*72*2  =  9216  } phase 2
#define OFF_SIM   64512                    // 64*65*4  = 16640  }
#define OFF_OUTS  46080                    // 256*68*4 = 69632 (aliases the above)
#define SMEM_SZ   115712

__global__ __launch_bounds__(256) void sapa_attn_kernel(
    const _Float16* __restrict__ qbuf,  // (B*128*128, 64) f16
    const _Float16* __restrict__ kbuf,  // (B*64*64, 64) f16
    const float*    __restrict__ xbuf,  // (B, 256, 64, 64) f32 (values)
    float*          __restrict__ out)   // (B, 256, 128, 128) f32
{
    __shared__ __align__(16) unsigned char smem[SMEM_SZ];
    _Float16* vtl  = (_Float16*)(smem + OFF_VTL);   // union values^T [c][pix]
    _Float16* attn = (_Float16*)(smem + OFF_ATTN);  // [q][pix] f16
    _Float16* qls  = (_Float16*)(smem + OFF_QLS);   // queries [q][e]
    _Float16* kls  = (_Float16*)(smem + OFF_KLS);   // union keys [pix][e]
    float*    sim  = (float*)   (smem + OFF_SIM);   // [q][pix] f32
    float*    outs = (float*)   (smem + OFF_OUTS);  // out stage [c][q] f32

    const int t    = threadIdx.x;
    const int wave = t >> 5;
    const int l    = t & 31;
    const int hi   = (l >= 16);

    const int blk = blockIdx.x;
    const int b   = blk >> 8;
    const int ty  = (blk >> 4) & 15, tx = blk & 15;
    const int cy0 = ty * 4, cx0 = tx * 4;    // low-res cell-tile origin

    // --- phase 0a: queries (64 q x 64 e), 16 halfs per thread ---
    {
        int qi = t >> 2, e0 = (t & 3) * 16;
        int ci = qi >> 2, sub = qi & 3;
        int cr = ci >> 2, cc = ci & 3;
        int hup = (cy0 + cr) * 2 + (sub >> 1);
        int wup = (cx0 + cc) * 2 + (sub & 1);
        const _Float16* qp = qbuf + (size_t)(b * 16384 + hup * 128 + wup) * 64 + e0;
        #pragma unroll
        for (int j = 0; j < 16; ++j) qls[qi * 72 + e0 + j] = qp[j];
    }
    // --- phase 0b: union keys (64 pix x 64 e), zero OOB ---
    {
        int u = t >> 2, e0 = (t & 3) * 16;
        int hy = cy0 - 2 + (u >> 3);
        int wx = cx0 - 2 + (u & 7);
        if (hy >= 0 && hy < 64 && wx >= 0 && wx < 64) {
            const _Float16* kp = kbuf + (size_t)(b * 4096 + hy * 64 + wx) * 64 + e0;
            #pragma unroll
            for (int j = 0; j < 16; ++j) kls[u * 72 + e0 + j] = kp[j];
        } else {
            #pragma unroll
            for (int j = 0; j < 16; ++j) kls[u * 72 + e0 + j] = (_Float16)0.0f;
        }
    }
    // --- phase 0c: union values transposed [c][pix]; NCHW rows load naturally ---
    {
        int cbase = t >> 4;               // 16 channels per pass
        int chunk = t & 15;               // 16 chunks of 4 pixels = 64 pixels
        int uy = chunk >> 1, h4 = chunk & 1;
        int hy = cy0 - 2 + uy;
        int wx0 = cx0 - 2 + h4 * 4;
        #pragma unroll 2
        for (int co = 0; co < 16; ++co) {
            int c = co * 16 + cbase;
            const float* xp = xbuf + ((size_t)(b * 256 + c) * 64 + hy) * 64;
            #pragma unroll
            for (int j = 0; j < 4; ++j) {
                int wx = wx0 + j;
                float v = 0.0f;
                if (hy >= 0 && hy < 64 && wx >= 0 && wx < 64) v = xp[wx];
                vtl[c * 72 + uy * 8 + h4 * 4 + j] = (_Float16)v;
            }
        }
    }
    __syncthreads();

    // --- phase 1: sim = Q @ Kunion^T  (M=64, N=64, K=64); 2 tile-pairs/wave ---
    {
        const int mt = wave >> 1;
        const int m  = l & 15;
        const int ko = hi ? 16 : 0, bo = hi ? 16 : 0;
        v16h a0, a1;
        const _Float16* qr = &qls[(mt * 16 + m) * 72];
        #pragma unroll
        for (int j = 0; j < 8; ++j) {
            a0[j] = qr[ko + j];      a0[j + 8] = qr[8 + ko + j];
            a1[j] = qr[32 + ko + j]; a1[j + 8] = qr[40 + ko + j];
        }
        #pragma unroll
        for (int i = 0; i < 2; ++i) {
            int nt = (wave * 2 + i) & 3;
            int n  = nt * 16 + (l & 15);
            const _Float16* kr = &kls[n * 72];
            v16h b0, b1;
            #pragma unroll
            for (int j = 0; j < 16; ++j) { b0[j] = kr[bo + j]; b1[j] = kr[32 + bo + j]; }
            v8f acc = {};
            acc = __builtin_amdgcn_wmma_f32_16x16x32_f16(false, a0, false, b0, (short)0, acc, false, false);
            acc = __builtin_amdgcn_wmma_f32_16x16x32_f16(false, a1, false, b1, (short)0, acc, false, false);
            #pragma unroll
            for (int r = 0; r < 8; ++r)
                sim[(mt * 16 + r + (hi ? 8 : 0)) * 65 + n] = acc[r];
        }
    }
    __syncthreads();

    // --- phase 2: per-query masked softmax over its 25-tap window ---
    if (t < 64) {
        int qi = t;
        int ci = qi >> 2;
        int cr = ci >> 2, cc = ci & 3;
        #pragma unroll
        for (int j = 0; j < 64; ++j) attn[qi * 72 + j] = (_Float16)0.0f;
        float s[25];
        float mx = -1e30f;
        #pragma unroll
        for (int dy = 0; dy < 5; ++dy)
            #pragma unroll
            for (int dx = 0; dx < 5; ++dx) {
                float v = sim[qi * 65 + (cr + dy) * 8 + (cc + dx)];
                s[dy * 5 + dx] = v;
                mx = fmaxf(mx, v);
            }
        float ssum = 0.0f;
        #pragma unroll
        for (int i = 0; i < 25; ++i) { s[i] = __expf(s[i] - mx); ssum += s[i]; }
        float inv = 1.0f / ssum;
        #pragma unroll
        for (int dy = 0; dy < 5; ++dy)
            #pragma unroll
            for (int dx = 0; dx < 5; ++dx)
                attn[qi * 72 + (cr + dy) * 8 + (cc + dx)] =
                    (_Float16)(s[dy * 5 + dx] * inv);
    }
    __syncthreads();   // sim/qls/kls dead from here; outs may reuse their LDS

    // --- phase 3: out = attn @ Vunion (M=64, N=256, K=64); D -> LDS stage ---
    {
        const int mt = wave >> 1;
        const int m  = l & 15;
        const int ko = hi ? 16 : 0, bo = hi ? 16 : 0;
        v16h a0, a1;
        const _Float16* ar = &attn[(mt * 16 + m) * 72];
        #pragma unroll
        for (int j = 0; j < 8; ++j) {
            a0[j] = ar[ko + j];      a0[j + 8] = ar[8 + ko + j];
            a1[j] = ar[32 + ko + j]; a1[j + 8] = ar[40 + ko + j];
        }
        #pragma unroll
        for (int i = 0; i < 8; ++i) {
            int nt = (wave * 8 + i) & 15;
            int c  = nt * 16 + (l & 15);
            const _Float16* vr = &vtl[c * 72];
            v16h b0, b1;
            #pragma unroll
            for (int j = 0; j < 16; ++j) { b0[j] = vr[bo + j]; b1[j] = vr[32 + bo + j]; }
            v8f acc = {};
            acc = __builtin_amdgcn_wmma_f32_16x16x32_f16(false, a0, false, b0, (short)0, acc, false, false);
            acc = __builtin_amdgcn_wmma_f32_16x16x32_f16(false, a1, false, b1, (short)0, acc, false, false);
            #pragma unroll
            for (int r = 0; r < 8; ++r) {
                int qi = mt * 16 + r + (hi ? 8 : 0);
                outs[c * 68 + qi] = acc[r];   // transpose stage: [channel][query]
            }
        }
    }
    __syncthreads();

    // --- epilogue: coalesced NCHW stores. 4096 float4 rows; consecutive lane
    //     pairs write the two 16B halves of one 8-float (c, h_up) row. ---
    #pragma unroll 4
    for (int i = 0; i < 16; ++i) {
        int fid  = i * 256 + t;
        int c    = fid >> 4;
        int sub  = (fid >> 1) & 7;         // (cell-row, p_y)
        int half = fid & 1;
        int cr = sub >> 1, py = sub & 1;
        int hup  = (cy0 + cr) * 2 + py;
        int wup0 = cx0 * 2 + half * 4;
        int base = cr * 16 + py * 2 + half * 8;   // query index of first elem
        float4 v;
        v.x = outs[c * 68 + base + 0];     // (cc=0|2, px=0)
        v.y = outs[c * 68 + base + 1];     // (cc=0|2, px=1)
        v.z = outs[c * 68 + base + 4];     // (cc=1|3, px=0)
        v.w = outs[c * 68 + base + 5];     // (cc=1|3, px=1)
        *(float4*)&out[(((size_t)b * 256 + c) * 128 + hup) * 128 + wup0] = v;
    }
}

// ---------------------------------------------------------------------------
extern "C" void kernel_launch(void* const* d_in, const int* in_sizes, int n_in,
                              void* d_out, int out_size, void* d_ws, size_t ws_size,
                              hipStream_t stream) {
    const float* y        = (const float*)d_in[0];
    const float* x        = (const float*)d_in[1];
    const float* norm_y_w = (const float*)d_in[2];
    const float* norm_y_b = (const float*)d_in[3];
    const float* norm_x_w = (const float*)d_in[4];
    const float* norm_x_b = (const float*)d_in[5];
    const float* q_w      = (const float*)d_in[6];
    const float* q_b      = (const float*)d_in[7];
    const float* k_w      = (const float*)d_in[8];
    const float* k_b      = (const float*)d_in[9];
    float* outp = (float*)d_out;

    // workspace: q f16 (65536 x 64) then k f16 (16384 x 64), ~10.5 MB
    _Float16* qbuf = (_Float16*)d_ws;
    _Float16* kbuf = qbuf + (size_t)65536 * 64;

    // q = LN(y) @ q_w^T + q_b : 65536 pixels / 64 per block
    ln_proj_kernel<<<1024, 128, 0, stream>>>(y, norm_y_w, norm_y_b, q_w, q_b,
                                             qbuf, 128 * 128);
    // k = LN(x) @ k_w^T + k_b : 16384 pixels / 64 per block
    ln_proj_kernel<<<256, 128, 0, stream>>>(x, norm_x_w, norm_x_b, k_w, k_b,
                                            kbuf, 64 * 64);
    // densified window attention, one block per 4x4-cell tile
    sapa_attn_kernel<<<1024, 256, 0, stream>>>(qbuf, kbuf, x, outp);
}
